// MolNet_49898930045492
// MI455X (gfx1250) — compile-verified, matrix operand order checked by reading, not run on hardware
//
#include <hip/hip_runtime.h>

// ---------------------------------------------------------------------------
// GINE GNN for MI455X (gfx1250, wave32).
// Node MLPs use v_wmma_f32_16x16x32_bf16 (bf16 in, f32 accumulate).
// Weights are pre-packed into per-fragment bf16 layout so each B fragment is
// one 32-byte load per lane. Edge gather/scatter via global_atomic_add_f32.
// ---------------------------------------------------------------------------

#define INC 32
#define HID 128

typedef __attribute__((ext_vector_type(16))) __bf16 bf16x16;
typedef __attribute__((ext_vector_type(8)))  __bf16 bf16x8;
typedef __attribute__((ext_vector_type(8)))  float  f32x8;

__device__ __forceinline__ float frelu(float v) { return v > 0.f ? v : 0.f; }

// K index for element i of a 16-bit A/B fragment (ISA 7.12.2, 16-bit 16x32):
// lanes 0-15 (kh=0): elems 0-7 -> K=0..7,  elems 8-15 -> K=16..23
// lanes 16-31 (kh=1): elems 0-7 -> K=8..15, elems 8-15 -> K=24..31
__device__ __forceinline__ int wmma_k(int i, int kh) {
  return (i < 8) ? (kh * 8 + i) : (8 + kh * 8 + i);
}

__device__ __forceinline__ f32x8 wmma_bf16(bf16x16 a, bf16x16 b, f32x8 c) {
  return __builtin_amdgcn_wmma_f32_16x16x32_bf16(
      /*neg_a=*/false, a, /*neg_b=*/false, b,
      /*c_mod=*/(short)0, c, /*reuse_a=*/false, /*reuse_b=*/false);
}

// Build an A fragment from two row-major f32 sources (summed), K base kbase.
// Elements map to two runs of 8 consecutive K values -> 4+4 float4 loads.
__device__ __forceinline__ bf16x16 make_afrag(const float* __restrict__ xr,
                                              const float* __restrict__ gr,
                                              int kbase, int kh) {
  const int s0 = kbase + kh * 8;
  const int s1 = kbase + 16 + kh * 8;
  float4 x0 = *(const float4*)(xr + s0);
  float4 x1 = *(const float4*)(xr + s0 + 4);
  float4 x2 = *(const float4*)(xr + s1);
  float4 x3 = *(const float4*)(xr + s1 + 4);
  float4 g0 = *(const float4*)(gr + s0);
  float4 g1 = *(const float4*)(gr + s0 + 4);
  float4 g2 = *(const float4*)(gr + s1);
  float4 g3 = *(const float4*)(gr + s1 + 4);
  bf16x16 a;
  a[0]  = (__bf16)(x0.x + g0.x); a[1]  = (__bf16)(x0.y + g0.y);
  a[2]  = (__bf16)(x0.z + g0.z); a[3]  = (__bf16)(x0.w + g0.w);
  a[4]  = (__bf16)(x1.x + g1.x); a[5]  = (__bf16)(x1.y + g1.y);
  a[6]  = (__bf16)(x1.z + g1.z); a[7]  = (__bf16)(x1.w + g1.w);
  a[8]  = (__bf16)(x2.x + g2.x); a[9]  = (__bf16)(x2.y + g2.y);
  a[10] = (__bf16)(x2.z + g2.z); a[11] = (__bf16)(x2.w + g2.w);
  a[12] = (__bf16)(x3.x + g3.x); a[13] = (__bf16)(x3.y + g3.y);
  a[14] = (__bf16)(x3.z + g3.z); a[15] = (__bf16)(x3.w + g3.w);
  return a;
}

// One 32B load per lane: fragment `frag`, packed layout [frag][lane][16 bf16].
__device__ __forceinline__ bf16x16 load_packed_frag(const __bf16* __restrict__ p,
                                                    int frag, int lane) {
  return *(const bf16x16*)(p + ((size_t)frag * 32 + lane) * 16);
}

// ---------------------------------------------------------------------------
// Weight pre-pack: row-major f32 W[K x ldn] -> bf16 fragments.
// frag id = kb * Nblocks + nb; one thread per (frag, lane) writes 16 bf16.
// ---------------------------------------------------------------------------
__global__ void __launch_bounds__(256) pack_weights(
    const float* __restrict__ W, __bf16* __restrict__ out,
    int Kblocks, int Nblocks, int ldn) {
  int tid = blockIdx.x * blockDim.x + threadIdx.x;
  if (tid >= Kblocks * Nblocks * 32) return;
  int lane = tid & 31;
  int frag = tid >> 5;
  int nb = frag % Nblocks;
  int kb = frag / Nblocks;
  int m  = lane & 15;
  int kh = lane >> 4;
  int n  = nb * 16 + m;
  __bf16* o = out + ((size_t)frag * 32 + lane) * 16;
#pragma unroll
  for (int i = 0; i < 16; ++i) {
    int k = kb * 32 + wmma_k(i, kh);
    o[i] = (__bf16)W[(size_t)k * ldn + n];
  }
}

// ---------------------------------------------------------------------------
// Edge kernel: msg = relu(x[src] + edge_attr @ We + be); atomic scatter to dst.
// One thread handles 4 channels of one edge (C = 32 or 128).
// ---------------------------------------------------------------------------
template <int C>
__global__ void __launch_bounds__(256) gine_edge(
    const float* __restrict__ xin, const int* __restrict__ src,
    const int* __restrict__ dst, const float* __restrict__ ea,
    const float* __restrict__ We, const float* __restrict__ be,
    float* __restrict__ agg, int nEdges) {
  constexpr int G = C / 4;
  int tid = blockIdx.x * blockDim.x + threadIdx.x;
  if (tid >= nEdges * G) return;
  int e  = tid / G;
  int c0 = (tid % G) * 4;
  int s = src[e];
  int d = dst[e];
  float a0 = ea[(size_t)e * 3 + 0];
  float a1 = ea[(size_t)e * 3 + 1];
  float a2 = ea[(size_t)e * 3 + 2];
  float4 xv = *(const float4*)(xin + (size_t)s * C + c0);
  float4 w0 = *(const float4*)(We + 0 * C + c0);
  float4 w1 = *(const float4*)(We + 1 * C + c0);
  float4 w2 = *(const float4*)(We + 2 * C + c0);
  float4 bv = *(const float4*)(be + c0);
  float m0 = frelu(xv.x + a0 * w0.x + a1 * w1.x + a2 * w2.x + bv.x);
  float m1 = frelu(xv.y + a0 * w0.y + a1 * w1.y + a2 * w2.y + bv.y);
  float m2 = frelu(xv.z + a0 * w0.z + a1 * w1.z + a2 * w2.z + bv.z);
  float m3 = frelu(xv.w + a0 * w0.w + a1 * w1.w + a2 * w2.w + bv.w);
  float* ap = agg + (size_t)d * C + c0;
  atomicAdd(ap + 0, m0);
  atomicAdd(ap + 1, m1);
  atomicAdd(ap + 2, m2);
  atomicAdd(ap + 3, m3);
}

// ---------------------------------------------------------------------------
// Layer-1 node MLP: h1 = relu( relu((x+agg1) @ W1a + b1a) @ W1b + b1b )
// One wave per 16-node tile. GEMM1: K=32 (8 WMMA). GEMM2: K=128 (32 WMMA).
// Intermediate transposed through a per-wave LDS tile (bf16, 16B-aligned).
// ---------------------------------------------------------------------------
__global__ void __launch_bounds__(256) gine_node1(
    const float* __restrict__ x, const float* __restrict__ agg,
    const __bf16* __restrict__ pWa, const float* __restrict__ ba,
    const __bf16* __restrict__ pWb, const float* __restrict__ bb,
    float* __restrict__ hout, int nNodes) {
  __shared__ alignas(16) __bf16 tbuf[8][16][HID + 8];  // pitch 272B (16B mult)
  const int lane = threadIdx.x & 31;
  const int wid  = threadIdx.x >> 5;
  const int base = (blockIdx.x * 8 + wid) * 16;
  if (base >= nNodes) return;  // uniform per wave
  const int m  = lane & 15;
  const int kh = lane >> 4;
  int row = base + m;
  if (row >= nNodes) row = nNodes - 1;

  const float* xr = x   + (size_t)row * INC;
  const float* gr = agg + (size_t)row * INC;
  const bf16x16 afrag = make_afrag(xr, gr, 0, kh);

  // GEMM1 (16x32)x(32x128) + bias + relu -> LDS (A layout for GEMM2)
#pragma unroll
  for (int nb = 0; nb < 8; ++nb) {
    const int n = nb * 16 + m;
    const bf16x16 bfrag = load_packed_frag(pWa, nb, lane);
    f32x8 acc = {};
    acc = wmma_bf16(afrag, bfrag, acc);
    const float bias = ba[n];
#pragma unroll
    for (int r = 0; r < 8; ++r)
      tbuf[wid][r + 8 * kh][n] = (__bf16)frelu(acc[r] + bias);
  }
  __asm__ volatile("s_wait_dscnt 0" ::: "memory");

  bf16x16 a2[4];
#pragma unroll
  for (int kb = 0; kb < 4; ++kb) {
    bf16x8 lo = *(const bf16x8*)&tbuf[wid][m][kb * 32 + kh * 8];
    bf16x8 hi = *(const bf16x8*)&tbuf[wid][m][kb * 32 + 16 + kh * 8];
    a2[kb] = __builtin_shufflevector(lo, hi, 0, 1, 2, 3, 4, 5, 6, 7,
                                     8, 9, 10, 11, 12, 13, 14, 15);
  }

  // GEMM2 (16x128)x(128x128) + bias + relu -> h1
#pragma unroll
  for (int nb = 0; nb < 8; ++nb) {
    const int n = nb * 16 + m;
    f32x8 acc = {};
#pragma unroll
    for (int kb = 0; kb < 4; ++kb) {
      const bf16x16 bfrag = load_packed_frag(pWb, kb * 8 + nb, lane);
      acc = wmma_bf16(a2[kb], bfrag, acc);
    }
    const float bias = bb[n];
#pragma unroll
    for (int r = 0; r < 8; ++r) {
      int nrow = base + r + 8 * kh;
      if (nrow < nNodes)
        hout[(size_t)nrow * HID + n] = frelu(acc[r] + bias);
    }
  }
}

// ---------------------------------------------------------------------------
// Layer-2 node MLP + fused global_add_pool:
//   h2 = relu( relu((h1+agg2) @ W2a + b2a) @ W2b + b2b )
//   pooled[batch[node]] += h2[node]       (f32 atomics)
// ---------------------------------------------------------------------------
__global__ void __launch_bounds__(256) gine_node2(
    const float* __restrict__ hin, const float* __restrict__ agg,
    const __bf16* __restrict__ pWa, const float* __restrict__ ba,
    const __bf16* __restrict__ pWb, const float* __restrict__ bb,
    const int* __restrict__ batch, float* __restrict__ pooled, int nNodes) {
  __shared__ alignas(16) __bf16 tbuf[8][16][HID + 8];
  const int lane = threadIdx.x & 31;
  const int wid  = threadIdx.x >> 5;
  const int base = (blockIdx.x * 8 + wid) * 16;
  if (base >= nNodes) return;
  const int m  = lane & 15;
  const int kh = lane >> 4;
  int row = base + m;
  if (row >= nNodes) row = nNodes - 1;

  const float* xr = hin + (size_t)row * HID;
  const float* gr = agg + (size_t)row * HID;
  bf16x16 afrag[4];
#pragma unroll
  for (int kb = 0; kb < 4; ++kb) afrag[kb] = make_afrag(xr, gr, kb * 32, kh);

  // GEMM1 (16x128)x(128x128) + bias + relu -> LDS
#pragma unroll
  for (int nb = 0; nb < 8; ++nb) {
    const int n = nb * 16 + m;
    f32x8 acc = {};
#pragma unroll
    for (int kb = 0; kb < 4; ++kb) {
      const bf16x16 bfrag = load_packed_frag(pWa, kb * 8 + nb, lane);
      acc = wmma_bf16(afrag[kb], bfrag, acc);
    }
    const float bias = ba[n];
#pragma unroll
    for (int r = 0; r < 8; ++r)
      tbuf[wid][r + 8 * kh][n] = (__bf16)frelu(acc[r] + bias);
  }
  __asm__ volatile("s_wait_dscnt 0" ::: "memory");

  bf16x16 a2[4];
#pragma unroll
  for (int kb = 0; kb < 4; ++kb) {
    bf16x8 lo = *(const bf16x8*)&tbuf[wid][m][kb * 32 + kh * 8];
    bf16x8 hi = *(const bf16x8*)&tbuf[wid][m][kb * 32 + 16 + kh * 8];
    a2[kb] = __builtin_shufflevector(lo, hi, 0, 1, 2, 3, 4, 5, 6, 7,
                                     8, 9, 10, 11, 12, 13, 14, 15);
  }

  // GEMM2 + bias + relu + fused pooling
#pragma unroll
  for (int nb = 0; nb < 8; ++nb) {
    const int n = nb * 16 + m;
    f32x8 acc = {};
#pragma unroll
    for (int kb = 0; kb < 4; ++kb) {
      const bf16x16 bfrag = load_packed_frag(pWb, kb * 8 + nb, lane);
      acc = wmma_bf16(a2[kb], bfrag, acc);
    }
    const float bias = bb[n];
#pragma unroll
    for (int r = 0; r < 8; ++r) {
      int nrow = base + r + 8 * kh;
      if (nrow < nNodes) {
        float v = frelu(acc[r] + bias);
        atomicAdd(pooled + (size_t)batch[nrow] * HID + n, v);
      }
    }
  }
}

// ---------------------------------------------------------------------------
// Readout: out[g] = relu(pooled[g] @ Wm1 + bm1) @ Wm2 + bm2  (tiny)
// ---------------------------------------------------------------------------
__global__ void __launch_bounds__(128) gine_readout(
    const float* __restrict__ pooled, const float* __restrict__ Wm1,
    const float* __restrict__ bm1, const float* __restrict__ Wm2,
    const float* __restrict__ bm2, float* __restrict__ out) {
  __shared__ float p[HID];
  __shared__ float red[HID];
  const int g = blockIdx.x;
  const int t = threadIdx.x;
  p[t] = pooled[(size_t)g * HID + t];
  __syncthreads();
  float acc = bm1[t];
#pragma unroll 8
  for (int k = 0; k < HID; ++k) acc += p[k] * Wm1[k * HID + t];
  red[t] = frelu(acc) * Wm2[t];
  __syncthreads();
  for (int s = 64; s > 0; s >>= 1) {
    if (t < s) red[t] += red[t + s];
    __syncthreads();
  }
  if (t == 0) out[g] = red[0] + bm2[0];
}

// ---------------------------------------------------------------------------
// Host-side launcher
// ---------------------------------------------------------------------------
extern "C" void kernel_launch(void* const* d_in, const int* in_sizes, int n_in,
                              void* d_out, int out_size, void* d_ws, size_t ws_size,
                              hipStream_t stream) {
  (void)n_in; (void)ws_size;
  const float* x     = (const float*)d_in[0];
  const int*   eidx  = (const int*)d_in[1];
  const float* ea    = (const float*)d_in[2];
  const int*   batch = (const int*)d_in[3];
  const float* We1 = (const float*)d_in[4],  *be1 = (const float*)d_in[5];
  const float* W1a = (const float*)d_in[6],  *b1a = (const float*)d_in[7];
  const float* W1b = (const float*)d_in[8],  *b1b = (const float*)d_in[9];
  const float* We2 = (const float*)d_in[10], *be2 = (const float*)d_in[11];
  const float* W2a = (const float*)d_in[12], *b2a = (const float*)d_in[13];
  const float* W2b = (const float*)d_in[14], *b2b = (const float*)d_in[15];
  const float* Wm1 = (const float*)d_in[16], *bm1 = (const float*)d_in[17];
  const float* Wm2 = (const float*)d_in[18], *bm2 = (const float*)d_in[19];

  const int N = in_sizes[0] / INC;   // 100000
  const int E = in_sizes[1] / 2;     // 1600000
  const int G = out_size;            // 2048 (OUT_C == 1)

  const int* src = eidx;
  const int* dst = eidx + E;

  // Workspace carve-up (offsets kept 256B-aligned)
  char* ws = (char*)d_ws;
  size_t off = 0;
  auto take = [&](size_t bytes) {
    void* p = ws + off;
    off += (bytes + 255) & ~(size_t)255;
    return p;
  };
  float*  agg1   = (float*)take((size_t)N * INC * sizeof(float));
  float*  h1     = (float*)take((size_t)N * HID * sizeof(float));
  float*  agg2   = (float*)take((size_t)N * HID * sizeof(float));
  float*  pooled = (float*)take((size_t)G * HID * sizeof(float));
  __bf16* p1a    = (__bf16*)take((size_t)1 * 8 * 512 * sizeof(__bf16));
  __bf16* p1b    = (__bf16*)take((size_t)4 * 8 * 512 * sizeof(__bf16));
  __bf16* p2a    = (__bf16*)take((size_t)4 * 8 * 512 * sizeof(__bf16));
  __bf16* p2b    = (__bf16*)take((size_t)4 * 8 * 512 * sizeof(__bf16));

  hipMemsetAsync(agg1,   0, (size_t)N * INC * sizeof(float), stream);
  hipMemsetAsync(agg2,   0, (size_t)N * HID * sizeof(float), stream);
  hipMemsetAsync(pooled, 0, (size_t)G * HID * sizeof(float), stream);

  // Pre-pack weights into WMMA fragment layout (bf16)
  pack_weights<<<1, 256, 0, stream>>>(W1a, p1a, 1, 8, HID);
  pack_weights<<<4, 256, 0, stream>>>(W1b, p1b, 4, 8, HID);
  pack_weights<<<4, 256, 0, stream>>>(W2a, p2a, 4, 8, HID);
  pack_weights<<<4, 256, 0, stream>>>(W2b, p2b, 4, 8, HID);

  // Layer 1
  {
    int total = E * (INC / 4);
    gine_edge<INC><<<(total + 255) / 256, 256, 0, stream>>>(
        x, src, dst, ea, We1, be1, agg1, E);
  }
  {
    int tiles = (N + 15) / 16;
    gine_node1<<<(tiles + 7) / 8, 256, 0, stream>>>(
        x, agg1, p1a, b1a, p1b, b1b, h1, N);
  }

  // Layer 2 (pooling fused)
  {
    int total = E * (HID / 4);
    gine_edge<HID><<<(total + 255) / 256, 256, 0, stream>>>(
        h1, src, dst, ea, We2, be2, agg2, E);
  }
  {
    int tiles = (N + 15) / 16;
    gine_node2<<<(tiles + 7) / 8, 256, 0, stream>>>(
        h1, agg2, p2a, b2a, p2b, b2b, batch, pooled, N);
  }

  // Readout
  gine_readout<<<G, 128, 0, stream>>>(pooled, Wm1, bm1, Wm2, bm2, (float*)d_out);
}
